// ModernMultiHeadAttention_21500606284349
// MI455X (gfx1250) — matrix operand order repaired and checked by reading, not compiled, not run on hardware
//
#include <hip/hip_runtime.h>
#include <cstdint>
#include <cstddef>

// MI455X / gfx1250, wave32. Matrix math via v_wmma_f32_16x16x32_f16 (f32 accum).
// All GEMMs consume B as B^T (N x K row-major) so both A- and B-fragments are
// contiguous 32B runs in LDS -> pure b128 staging, no element scatter.
// Tile staging uses GLOBAL_LOAD_ASYNC_TO_LDS_B128 (ASYNCcnt) when available.

typedef __attribute__((ext_vector_type(16))) _Float16 v16h;
typedef __attribute__((ext_vector_type(8)))  float    v8f;

#define BQ   4
#define SQ   2048
#define DQ   2048
#define HQ   16
#define DKQ  128
#define NEGV (-1e9f)

// ---------------------------------------------------------------------------
// Async global->LDS staging. Builtin signature (from hipcc diagnostic):
//   (AS1 int __vector_size__(16)*, AS3 int __vector_size__(16)*, imm offset, imm cpol)
// ---------------------------------------------------------------------------
#if defined(__has_builtin)
#if __has_builtin(__builtin_amdgcn_global_load_async_to_lds_b128)
#define HAVE_ASYNC_LDS 1
#endif
#endif

typedef int b128v __attribute__((__vector_size__(4 * sizeof(int))));
typedef __attribute__((address_space(1))) b128v* gptr_b128;
typedef __attribute__((address_space(3))) b128v* lptr_b128;

__device__ __forceinline__ void stage16B(const void* g, void* l) {
#ifdef HAVE_ASYNC_LDS
  __builtin_amdgcn_global_load_async_to_lds_b128(
      (gptr_b128)g, (lptr_b128)l, /*offset=*/0, /*cpol=*/0);
#else
  *(float4*)l = *(const float4*)g;
#endif
}

__device__ __forceinline__ void stage_wait() {
#ifdef HAVE_ASYNC_LDS
  asm volatile("s_wait_asynccnt 0x0" ::: "memory");
#endif
}

// ---------------------------------------------------------------------------
// Conversion kernel (f32 -> f16 copy; no transposes needed anywhere)
// ---------------------------------------------------------------------------
__global__ __launch_bounds__(256) void k_f32_to_f16(const float* __restrict__ in,
                                                    _Float16* __restrict__ out,
                                                    size_t n) {
  size_t i = (size_t)blockIdx.x * 256 + threadIdx.x;
  if (i < n) out[i] = (_Float16)in[i];
}

// ---------------------------------------------------------------------------
// WMMA GEMM core: C(128x128 block) = A[M,K] * Bt[N,K]^T, f32 accumulate.
// A: row-major (f16, or f32 converted while staging). Bt: row-major f16 N x K.
// Block = 256 threads = 8 wave32; wave w: rows (w&3)*32..+31, cols (w>>2)*64..+63
// => 2x4 v8f accumulators per wave.
// Fragment layouts used here (A and B share contiguous k; C per ISA 7.12.2):
//   A elem i : row = lo,  K = 16*hi + i                             (one 32B run)
//   B elem i : N  = lo,  K = 16*hi + i                             (one 32B run)
//   C elem r : M = 8*hi + r, N = lo
// ---------------------------------------------------------------------------
template <typename AElem>
__device__ __forceinline__ void gemm_core(const AElem* __restrict__ Ag, size_t lda,
                                          const _Float16* __restrict__ Btg, size_t ldbt,
                                          int K, size_t m_blk, size_t n_blk,
                                          v8f acc[2][4]) {
  constexpr int P = 40;                  // LDS pitch (f16): 80B rows, 16B aligned
  __shared__ _Float16 As[128 * P];       // 10 KB
  __shared__ _Float16 Bs[128 * P];       // 10 KB

  const int tid  = threadIdx.x;
  const int lane = tid & 31, lo = lane & 15, hi = lane >> 4;
  const int wave = tid >> 5;
  const int wm = (wave & 3) * 32, wn = (wave >> 2) * 64;
  const int r0 = tid >> 1;               // 0..127 : staging row (A and Bt)
  const int c0 = (tid & 1) * 16;         // 0 / 16 : staging col chunk (16 f16)

  const AElem*    aSrc = Ag  + (m_blk + (size_t)r0) * lda  + (size_t)c0;
  const _Float16* bSrc = Btg + (n_blk + (size_t)r0) * ldbt + (size_t)c0;
  _Float16* aDst = &As[r0 * P + c0];
  _Float16* bDst = &Bs[r0 * P + c0];

  for (int k0 = 0; k0 < K; k0 += 32) {
    // ---- stage A tile (128 x 32) ----
    if constexpr (sizeof(AElem) == 2) {
      stage16B(aSrc + k0,     aDst);
      stage16B(aSrc + k0 + 8, aDst + 8);
    } else {
      // f32 probs -> f16 while staging (P@V GEMM)
#pragma unroll
      for (int i = 0; i < 16; ++i) aDst[i] = (_Float16)aSrc[k0 + i];
    }
    // ---- stage Bt tile (128 x 32) ----
    stage16B(bSrc + k0,     bDst);
    stage16B(bSrc + k0 + 8, bDst + 8);

    stage_wait();
    __syncthreads();

    if (k0 + 32 < K) {                   // global_prefetch_b8 for next K-tile
      __builtin_prefetch(aSrc + k0 + 32, 0, 0);
      __builtin_prefetch(bSrc + k0 + 32, 0, 0);
    }

    // ---- compute: 2x4 WMMAs per wave ----
#pragma unroll
    for (int mi = 0; mi < 2; ++mi) {
      const _Float16* rowp = &As[(wm + mi * 16 + lo) * P];
      v16h a;
#pragma unroll
      for (int i = 0; i < 8; ++i) a[i] = rowp[16 * hi + i];
#pragma unroll
      for (int i = 0; i < 8; ++i) a[8 + i] = rowp[16 * hi + 8 + i];
#pragma unroll
      for (int ni = 0; ni < 4; ++ni) {
        const _Float16* bp = &Bs[(wn + ni * 16 + lo) * P + 16 * hi];
        v16h b;
#pragma unroll
        for (int i = 0; i < 16; ++i) b[i] = bp[i];
        acc[mi][ni] = __builtin_amdgcn_wmma_f32_16x16x32_f16(
            false, a, false, b, (short)0, acc[mi][ni], false, false);
      }
    }
    __syncthreads();
  }
}

__device__ __forceinline__ void zero_acc(v8f acc[2][4]) {
  v8f z = {0.f, 0.f, 0.f, 0.f, 0.f, 0.f, 0.f, 0.f};
#pragma unroll
  for (int i = 0; i < 2; ++i)
#pragma unroll
    for (int j = 0; j < 4; ++j) acc[i][j] = z;
}

// ---------------------------------------------------------------------------
// GEMM 1: qkv = x @ w_qkv^T  (8192 x 6144); Bt = w_qkv itself (f16 copy).
// Epilogue: RoPE on Q (with 1/sqrt(dk) folded) and K; V scattered transposed.
// ---------------------------------------------------------------------------
__global__ __launch_bounds__(256) void k_gemm_qkv(const _Float16* __restrict__ xh,
                                                  const _Float16* __restrict__ wqkvh,
                                                  _Float16* __restrict__ qh,
                                                  _Float16* __restrict__ kh,
                                                  _Float16* __restrict__ vhT) {
  const size_t m_blk = (size_t)blockIdx.y * 128;
  const size_t n_blk = (size_t)blockIdx.x * 128;
  v8f acc[2][4];
  zero_acc(acc);
  gemm_core<_Float16>(xh, DQ, wqkvh, DQ, DQ, m_blk, n_blk, acc);

  const int tid = threadIdx.x, lane = tid & 31, lo = lane & 15, hi = lane >> 4;
  const int wave = tid >> 5, wm = (wave & 3) * 32, wn = (wave >> 2) * 64;
  const float kfreq = -(logf(10000.0f) / 128.0f);

#pragma unroll
  for (int mi = 0; mi < 2; ++mi)
#pragma unroll
    for (int ni = 0; ni < 4; ++ni)
#pragma unroll
      for (int r = 0; r < 8; ++r) {
        const size_t row = m_blk + (size_t)(wm + mi * 16 + 8 * hi + r);
        const size_t col = n_blk + (size_t)(wn + ni * 16 + lo);
        const int b = (int)(row >> 11), s = (int)(row & 2047);
        const int which = (int)(col >> 11), d = (int)(col & 2047);
        const int h = d >> 7, dk = d & 127;
        const float v = acc[mi][ni][r];
        const size_t bh = (size_t)b * HQ + (size_t)h;
        if (which == 2) {
          vhT[(bh * DKQ + (size_t)dk) * SQ + (size_t)s] = (_Float16)v;
        } else {
          const float freq = __expf(kfreq * (float)(dk & ~1));
          const float ang = (float)s * freq;
          const float rf = __cosf(ang) + __sinf(ang);
          if (which == 0)
            qh[(bh * SQ + (size_t)s) * DKQ + (size_t)dk] =
                (_Float16)(v * rf * 0.08838834764831845f); // 1/sqrt(128) folded in
          else
            kh[(bh * SQ + (size_t)s) * DKQ + (size_t)dk] = (_Float16)(v * rf);
        }
      }
}

// ---------------------------------------------------------------------------
// GEMM 2: scores(b,h) = Q(b,h) @ K(b,h)^T ; Bt = K (row-major [S,DK]).
// Fused causal + padding mask -> f32 scores in d_out attn region.
// ---------------------------------------------------------------------------
__global__ __launch_bounds__(256) void k_gemm_scores(const _Float16* __restrict__ qh,
                                                     const _Float16* __restrict__ kh,
                                                     const float* __restrict__ amask,
                                                     float* __restrict__ attnP) {
  const int bh = blockIdx.z;
  const int b = bh >> 4;
  const size_t m_blk = (size_t)blockIdx.y * 128;
  const size_t n_blk = (size_t)blockIdx.x * 128;
  const _Float16* A  = qh + (size_t)bh * SQ * DKQ;
  const _Float16* Bt = kh + (size_t)bh * SQ * DKQ;
  v8f acc[2][4];
  zero_acc(acc);
  gemm_core<_Float16>(A, DKQ, Bt, DKQ, DKQ, m_blk, n_blk, acc);

  const int tid = threadIdx.x, lane = tid & 31, lo = lane & 15, hi = lane >> 4;
  const int wave = tid >> 5, wm = (wave & 3) * 32, wn = (wave >> 2) * 64;
  float* out = attnP + (size_t)bh * SQ * SQ;

#pragma unroll
  for (int mi = 0; mi < 2; ++mi)
#pragma unroll
    for (int ni = 0; ni < 4; ++ni)
#pragma unroll
      for (int r = 0; r < 8; ++r) {
        const size_t q = m_blk + (size_t)(wm + mi * 16 + 8 * hi + r);
        const size_t c = n_blk + (size_t)(wn + ni * 16 + lo);
        float sc = acc[mi][ni][r];
        if (c > q) sc = NEGV;                                  // causal
        sc += (1.0f - amask[(size_t)b * SQ + c]) * NEGV;       // padding mask
        out[q * SQ + c] = sc;
      }
}

// ---------------------------------------------------------------------------
// Row softmax in-place on the attn region (one block per row of 2048)
// ---------------------------------------------------------------------------
__global__ __launch_bounds__(256) void k_softmax_rows(float* __restrict__ attnP) {
  float* p = attnP + (size_t)blockIdx.x * SQ;
  const int tid = threadIdx.x;
  __shared__ float red[256];
  float vals[8];
  float m = -3.4e38f;
#pragma unroll
  for (int j = 0; j < 8; ++j) {
    vals[j] = p[tid + j * 256];
    m = fmaxf(m, vals[j]);
  }
  red[tid] = m;
  __syncthreads();
  for (int s = 128; s > 0; s >>= 1) {
    if (tid < s) red[tid] = fmaxf(red[tid], red[tid + s]);
    __syncthreads();
  }
  m = red[0];
  __syncthreads();
  float sum = 0.f;
#pragma unroll
  for (int j = 0; j < 8; ++j) {
    vals[j] = __expf(vals[j] - m);
    sum += vals[j];
  }
  red[tid] = sum;
  __syncthreads();
  for (int s = 128; s > 0; s >>= 1) {
    if (tid < s) red[tid] += red[tid + s];
    __syncthreads();
  }
  const float inv = 1.0f / red[0];
#pragma unroll
  for (int j = 0; j < 8; ++j) p[tid + j * 256] = vals[j] * inv;
}

// ---------------------------------------------------------------------------
// GEMM 3: out(b,h) = P(b,h) @ V(b,h) ; A = f32 probs (converted in staging),
// Bt = V^T (row-major [DK,S]).  Epilogue writes head-concat f16 [B,S,D].
// ---------------------------------------------------------------------------
__global__ __launch_bounds__(256) void k_gemm_av(const float* __restrict__ attnP,
                                                 const _Float16* __restrict__ vhT,
                                                 _Float16* __restrict__ aout) {
  const int bh = blockIdx.z;
  const int b = bh >> 4, h = bh & 15;
  const size_t m_blk = (size_t)blockIdx.y * 128;
  const float* A = attnP + (size_t)bh * SQ * SQ;
  const _Float16* Bt = vhT + (size_t)bh * DKQ * SQ;
  v8f acc[2][4];
  zero_acc(acc);
  gemm_core<float>(A, SQ, Bt, SQ, SQ, m_blk, 0, acc);

  const int tid = threadIdx.x, lane = tid & 31, lo = lane & 15, hi = lane >> 4;
  const int wave = tid >> 5, wm = (wave & 3) * 32, wn = (wave >> 2) * 64;

#pragma unroll
  for (int mi = 0; mi < 2; ++mi)
#pragma unroll
    for (int ni = 0; ni < 4; ++ni)
#pragma unroll
      for (int r = 0; r < 8; ++r) {
        const size_t s = m_blk + (size_t)(wm + mi * 16 + 8 * hi + r);
        const int dk = wn + ni * 16 + lo;
        aout[((size_t)b * SQ + s) * DQ + (size_t)(h * DKQ + dk)] =
            (_Float16)acc[mi][ni][r];
      }
}

// ---------------------------------------------------------------------------
// GEMM 4: output = aout @ w_o^T ; Bt = w_o itself (f16 copy) -> f32 d_out
// ---------------------------------------------------------------------------
__global__ __launch_bounds__(256) void k_gemm_out(const _Float16* __restrict__ aout,
                                                  const _Float16* __restrict__ woh,
                                                  float* __restrict__ outP) {
  const size_t m_blk = (size_t)blockIdx.y * 128;
  const size_t n_blk = (size_t)blockIdx.x * 128;
  v8f acc[2][4];
  zero_acc(acc);
  gemm_core<_Float16>(aout, DQ, woh, DQ, DQ, m_blk, n_blk, acc);

  const int tid = threadIdx.x, lane = tid & 31, lo = lane & 15, hi = lane >> 4;
  const int wave = tid >> 5, wm = (wave & 3) * 32, wn = (wave >> 2) * 64;

#pragma unroll
  for (int mi = 0; mi < 2; ++mi)
#pragma unroll
    for (int ni = 0; ni < 4; ++ni)
#pragma unroll
      for (int r = 0; r < 8; ++r) {
        const size_t row = m_blk + (size_t)(wm + mi * 16 + 8 * hi + r);
        const size_t col = n_blk + (size_t)(wn + ni * 16 + lo);
        outP[row * DQ + col] = acc[mi][ni][r];
      }
}

// ---------------------------------------------------------------------------
// Host launcher
// ---------------------------------------------------------------------------
extern "C" void kernel_launch(void* const* d_in, const int* in_sizes, int n_in,
                              void* d_out, int out_size, void* d_ws, size_t ws_size,
                              hipStream_t stream) {
  (void)in_sizes; (void)n_in; (void)out_size; (void)ws_size;

  const float* x     = (const float*)d_in[0];   // [B,S,D]
  const float* amask = (const float*)d_in[1];   // [B,S]
  const float* wqkv  = (const float*)d_in[2];   // [3D,D]
  const float* wo    = (const float*)d_in[3];   // [D,D]

  float* outP  = (float*)d_out;                            // [B,S,D]
  float* attnP = outP + (size_t)BQ * SQ * DQ;              // [B,H,S,S]

  // workspace carve-up (f16 buffers), ~192 MB total
  char* ws = (char*)d_ws;
  _Float16* xh    = (_Float16*)ws;  ws += (size_t)BQ * SQ * DQ * 2;          // 32 MB
  _Float16* wqkvh = (_Float16*)ws;  ws += (size_t)3 * DQ * DQ * 2;           // 24 MB
  _Float16* woh   = (_Float16*)ws;  ws += (size_t)DQ * DQ * 2;               //  8 MB
  _Float16* qh    = (_Float16*)ws;  ws += (size_t)BQ * HQ * SQ * DKQ * 2;    // 32 MB
  _Float16* kh    = (_Float16*)ws;  ws += (size_t)BQ * HQ * SQ * DKQ * 2;    // 32 MB
  _Float16* vhT   = (_Float16*)ws;  ws += (size_t)BQ * HQ * DKQ * SQ * 2;    // 32 MB
  _Float16* aout  = (_Float16*)ws;  ws += (size_t)BQ * SQ * DQ * 2;          // 32 MB

  // 1) precision staging (plain copies; B^T layouts are the native layouts)
  {
    size_t n = (size_t)BQ * SQ * DQ;
    k_f32_to_f16<<<(unsigned)((n + 255) / 256), 256, 0, stream>>>(x, xh, n);
  }
  {
    size_t n = (size_t)3 * DQ * DQ;
    k_f32_to_f16<<<(unsigned)((n + 255) / 256), 256, 0, stream>>>(wqkv, wqkvh, n);
  }
  {
    size_t n = (size_t)DQ * DQ;
    k_f32_to_f16<<<(unsigned)((n + 255) / 256), 256, 0, stream>>>(wo, woh, n);
  }

  // 2) QKV projection + RoPE
  k_gemm_qkv<<<dim3(3 * DQ / 128, BQ * SQ / 128), 256, 0, stream>>>(xh, wqkvh, qh, kh, vhT);

  // 3) scores + masks
  k_gemm_scores<<<dim3(SQ / 128, SQ / 128, BQ * HQ), 256, 0, stream>>>(qh, kh, amask, attnP);

  // 4) softmax in place (probs are the second returned output)
  k_softmax_rows<<<(unsigned)((size_t)BQ * HQ * SQ), 256, 0, stream>>>(attnP);

  // 5) P @ V
  k_gemm_av<<<dim3(1, SQ / 128, BQ * HQ), 256, 0, stream>>>(attnP, vhT, aout);

  // 6) output projection
  k_gemm_out<<<dim3(DQ / 128, BQ * SQ / 128), 256, 0, stream>>>(aout, woh, outP);
}
